// StructuralTypeSeqModel_81097572483758
// MI455X (gfx1250) — compile-verified, matrix operand order checked by reading, not run on hardware
//
#include <hip/hip_runtime.h>

// ---------------- problem constants (from reference) ----------------
#define N_NODES  50000
#define E_EDGES  800000
#define E_TOT    (E_EDGES + N_NODES)   // edges + self loops
#define D_IN     128
#define H_DIM    256
#define B_GRAPHS 64
#define SEQ_LEN  20
#define C_CLS    32
#define OUT_COLS (C_CLS * SEQ_LEN)     // 640
#define NEG_SLOPE 0.2f

#define GEMM_THREADS 512               // 16 wave32s: one 16x16 N-tile each
#define MAX_K   256
#define A_PAD   4                      // +4 floats/row: spreads LDS banks, keeps 16B align

typedef __attribute__((ext_vector_type(2)))  float        v2f;
typedef __attribute__((ext_vector_type(8)))  float        v8f;
typedef __attribute__((ext_vector_type(16))) _Float16     v16h;
typedef __attribute__((ext_vector_type(4)))  unsigned int u32x4;
typedef __attribute__((ext_vector_type(4)))  int          i32x4;
typedef __attribute__((ext_vector_type(8)))  int          i32x8;

// =====================================================================
// 1) Dense GEMM  H = X @ W  via WMMA.
//    Block = 16 waves covering one M-tile (16 rows) x all 256 cols.
//    A tile (16 x K) staged in LDS once per block:
//      - TDM tensor_load_to_lds with pad_enable (4 DWORDs per K-DWORD row)
//        when available; cooperative global loads otherwise.
//    M = 50000 (multiple of 16), K in {128,256}, N = 256.
// =====================================================================
__global__ __launch_bounds__(GEMM_THREADS)
void gat_gemm_wmma(const float* __restrict__ X, const float* __restrict__ W,
                   float* __restrict__ Hout, int M, int K, int N) {
  __shared__ float Atile[16 * (MAX_K + A_PAD)];
  const int STR  = K + A_PAD;                 // padded LDS row stride (floats)
  const int row0 = (int)blockIdx.x * 16;      // M-tile base row
  if (row0 >= M) return;

#if defined(__has_builtin) && __has_builtin(__builtin_amdgcn_tensor_load_to_lds)
  // ---- Tensor Data Mover: one wave issues the 2D tile DMA ----
  if (threadIdx.x < 32) {
    const unsigned long long ga = (unsigned long long)(const void*)(X + (long)row0 * K);
    const unsigned lds_off = (unsigned)(unsigned long long)(__attribute__((address_space(3))) void*)Atile;
    // D# group 0: count=1 | lds_addr | global_addr[56:0] | type=2
    u32x4 g0;
    g0[0] = 1u;                                          // count=1, user mode
    g0[1] = lds_off;                                     // lds_addr (bytes)
    g0[2] = (unsigned)(ga & 0xFFFFFFFFu);                // global_addr[31:0]
    g0[3] = (unsigned)((ga >> 32) & 0x01FFFFFFu) | 0x80000000u; // ga[56:32], type=2
    // D# group 1: data_size=4B, pad every K DWORDs by A_PAD DWORDs,
    // tensor = 16 x K, tile = 16 x K, dim0 stride = K
    const unsigned pad_interval = (K == 256) ? 7u : 6u;  // 256 / 128 DWORDs
    const unsigned pad_amount   = (unsigned)(A_PAD - 1); // 4 DWORDs
    const unsigned dim0 = (unsigned)K, dim1 = 16u;
    const unsigned til0 = (unsigned)K, til1 = 16u;
    i32x8 g1;
    g1[0] = (int)((2u << 16) | (1u << 20) | (pad_interval << 22) | (pad_amount << 25));
    g1[1] = (int)((dim0 & 0xFFFFu) << 16);                               // tensor_dim0 lo
    g1[2] = (int)(((dim0 >> 16) & 0xFFFFu) | ((dim1 & 0xFFFFu) << 16));  // dim0 hi | dim1 lo
    g1[3] = (int)(((dim1 >> 16) & 0xFFFFu) | ((til0 & 0xFFFFu) << 16));  // dim1 hi | tile_dim0
    g1[4] = (int)(til1 & 0xFFFFu);                                       // tile_dim1, tile_dim2=0
    g1[5] = (int)dim0;                                                   // dim0_stride lo32 = K
    g1[6] = 0;                                                           // stride hi | dim1_stride lo
    g1[7] = 0;
    i32x4 gz = {0, 0, 0, 0};
#if __clang_major__ >= 23
    i32x8 gz8 = {0, 0, 0, 0, 0, 0, 0, 0};
    __builtin_amdgcn_tensor_load_to_lds(g0, g1, gz, gz, gz8, 0);
#else
    __builtin_amdgcn_tensor_load_to_lds(g0, g1, gz, gz, 0);
#endif
    __builtin_amdgcn_s_wait_tensorcnt(0);
  }
#else
  // ---- fallback: cooperative coalesced staging ----
  for (int i = (int)threadIdx.x; i < 16 * K; i += GEMM_THREADS) {
    const int r = i / K, c = i - r * K;
    Atile[r * STR + c] = X[(long)(row0 + r) * K + c];
  }
#endif
  __syncthreads();

  const int lane = threadIdx.x & 31;
  const int tn   = (int)(threadIdx.x >> 5);   // N-tile (0..15)
  const int half = lane >> 4;                 // 0: lanes 0-15, 1: lanes 16-31
  const int l16  = lane & 15;
  const float* Wc    = W + (long)tn * 16 + l16;       // B column for this lane
  const float* Arow  = Atile + l16 * STR;             // A row (LDS) for this lane

  v8f acc = {};

#if defined(__has_builtin) && __has_builtin(__builtin_amdgcn_wmma_f32_16x16x4_f32)
  // Exact fp32 path: V_WMMA_F32_16X16X4_F32.
  // A 16x4: lane<16 holds M=l16, K={k,k+1}; lane>=16 K={k+2,k+3}. B mirrored.
  for (int k = 0; k < K; k += 4) {
    if ((k & 15) == 0)
      __builtin_prefetch(&Wc[(long)(k + 16) * N], 0, 0);   // global_prefetch_b8
    v2f a, b;
    a[0] = Arow[k + 2 * half + 0];
    a[1] = Arow[k + 2 * half + 1];
    b[0] = Wc[(long)(k + 2 * half + 0) * N];
    b[1] = Wc[(long)(k + 2 * half + 1) * N];
    acc = __builtin_amdgcn_wmma_f32_16x16x4_f32(false, a, false, b,
                                                (short)0, acc, false, false);
  }
#else
  // Fallback: codegen-confirmed V_WMMA_F32_16X16X32_F16 (fp32 accumulate).
  for (int k = 0; k < K; k += 32) {
    v16h a, b;
#pragma unroll
    for (int j = 0; j < 8; ++j) {
      a[j]     = (_Float16)Arow[k + 8 * half + j];
      a[j + 8] = (_Float16)Arow[k + 16 + 8 * half + j];
      b[j]     = (_Float16)Wc[(long)(k + 8 * half + j) * N];
      b[j + 8] = (_Float16)Wc[(long)(k + 16 + 8 * half + j) * N];
    }
    acc = __builtin_amdgcn_wmma_f32_16x16x32_f16(false, a, false, b,
                                                 (short)0, acc, false, false);
  }
#endif

  // C/D 16x16 f32 layout: VGPR v = row (row0 + v + 8*half), col = tn*16 + l16
  const long obase = (long)(row0 + 8 * half) * N + (long)tn * 16 + l16;
#pragma unroll
  for (int v = 0; v < 8; ++v)
    Hout[obase + (long)v * N] = acc[v];
}

// =====================================================================
// 2) Per-node attention logits (one wave per node, wave32 reduction)
// =====================================================================
__global__ __launch_bounds__(256)
void gat_node_alpha(const float* __restrict__ Hm, const float* __restrict__ a_s,
                    const float* __restrict__ a_d, float* __restrict__ as_o,
                    float* __restrict__ ad_o) {
  const int lane = threadIdx.x & 31;
  const int node = (int)((blockIdx.x * blockDim.x + threadIdx.x) >> 5);
  if (node >= N_NODES) return;
  const float* hrow = Hm + (long)node * H_DIM;
  float s = 0.f, d = 0.f;
#pragma unroll
  for (int j = 0; j < H_DIM / 32; ++j) {
    const int c = lane + 32 * j;
    const float hv = hrow[c];
    s += hv * a_s[c];
    d += hv * a_d[c];
  }
#pragma unroll
  for (int off = 16; off; off >>= 1) {
    s += __shfl_xor(s, off, 32);
    d += __shfl_xor(d, off, 32);
  }
  if (lane == 0) { as_o[node] = s; ad_o[node] = d; }
}

// =====================================================================
// 3) Segment-state init: agg = 0, m = -inf, denom = 0
// =====================================================================
__global__ __launch_bounds__(256)
void gat_init_seg(float* __restrict__ agg, float* __restrict__ mx,
                  float* __restrict__ den) {
  const long i = (long)blockIdx.x * blockDim.x + threadIdx.x;
  if (i < (long)N_NODES * H_DIM) agg[i] = 0.f;
  if (i < N_NODES) { mx[i] = -__builtin_inff(); den[i] = 0.f; }
}

// ---- sign-aware float atomic max (monotone over float bit patterns) ----
__device__ __forceinline__ void atomicMaxF(float* addr, float v) {
  if (v >= 0.f) atomicMax((int*)addr, __float_as_int(v));
  else          atomicMin((unsigned int*)addr, __float_as_uint(v));
}

__device__ __forceinline__ void edge_sd(const int* __restrict__ ei, int e,
                                        int& s, int& d) {
  if (e < E_EDGES) { s = ei[e]; d = ei[E_EDGES + e]; }
  else             { s = e - E_EDGES; d = s; }        // appended self loops
}

// =====================================================================
// 4) Edge pass 1: e = leaky_relu(as[src] + ad[dst]); segment max into m[dst]
// =====================================================================
__global__ __launch_bounds__(256)
void gat_edge_logit_max(const int* __restrict__ ei, const float* __restrict__ as_,
                        const float* __restrict__ ad_, float* __restrict__ evals,
                        float* __restrict__ mx) {
  const int e = (int)(blockIdx.x * blockDim.x + threadIdx.x);
  if (e >= E_TOT) return;
  int s, d; edge_sd(ei, e, s, d);
  float v = as_[s] + ad_[d];
  v = v > 0.f ? v : NEG_SLOPE * v;
  evals[e] = v;
  atomicMaxF(&mx[d], v);
}

// =====================================================================
// 5) Edge pass 2: ex = exp(e - m[dst]); segment sum into denom[dst]
// =====================================================================
__global__ __launch_bounds__(256)
void gat_edge_expsum(const int* __restrict__ ei, float* __restrict__ evals,
                     const float* __restrict__ mx, float* __restrict__ den) {
  const int e = (int)(blockIdx.x * blockDim.x + threadIdx.x);
  if (e >= E_TOT) return;
  int s, d; edge_sd(ei, e, s, d);
  const float ex = expf(evals[e] - mx[d]);
  evals[e] = ex;
  atomicAdd(&den[d], ex);
}

// =====================================================================
// 6) Edge pass 3: agg[dst] += (ex/denom[dst]) * h[src]
//    one wave per edge, 8 strided columns per lane (coalesced rows,
//    h is L2-resident: 51.2 MB << 192 MB)
// =====================================================================
__global__ __launch_bounds__(256)
void gat_edge_aggregate(const int* __restrict__ ei, const float* __restrict__ Hm,
                        const float* __restrict__ evals, const float* __restrict__ den,
                        float* __restrict__ agg) {
  const int lane = threadIdx.x & 31;
  const int e    = (int)((blockIdx.x * blockDim.x + threadIdx.x) >> 5);
  if (e >= E_TOT) return;
  int s, d; edge_sd(ei, e, s, d);
  const float w   = evals[e] / (den[d] + 1e-16f);
  const float* hs = Hm  + (long)s * H_DIM;
  float*       ag = agg + (long)d * H_DIM;
#pragma unroll
  for (int j = 0; j < H_DIM / 32; ++j) {
    const int c = lane + 32 * j;
    atomicAdd(&ag[c], w * hs[c]);
  }
}

// =====================================================================
// 7) Epilogue: out = relu(agg + b)
// =====================================================================
__global__ __launch_bounds__(256)
void gat_bias_relu(const float* __restrict__ agg, const float* __restrict__ b,
                   float* __restrict__ out) {
  const long i = (long)blockIdx.x * blockDim.x + threadIdx.x;
  if (i >= (long)N_NODES * H_DIM) return;
  const float v = agg[i] + b[(int)(i & (H_DIM - 1))];
  out[i] = v > 0.f ? v : 0.f;
}

// =====================================================================
// 8) node0[b] = lower_bound(batch, b)   (batch is sorted)
// =====================================================================
__global__ void gat_find_node0(const int* __restrict__ batch, int* __restrict__ node0) {
  const int b = (int)threadIdx.x;
  if (b >= B_GRAPHS) return;
  int lo = 0, hi = N_NODES;
  while (lo < hi) {
    const int mid = (lo + hi) >> 1;
    if (batch[mid] < b) lo = mid + 1; else hi = mid;
  }
  node0[b] = lo;
}

// =====================================================================
// 9) Projection: logits[b] = (agg[node0[b]] + b2) @ Wp + bp
// =====================================================================
__global__ __launch_bounds__(256)
void gat_project(const float* __restrict__ agg, const float* __restrict__ b2,
                 const int* __restrict__ node0, const float* __restrict__ Wp,
                 const float* __restrict__ bp, float* __restrict__ out) {
  __shared__ float in[H_DIM];
  const int g = (int)blockIdx.x;
  const int n = node0[g];
  for (int c = (int)threadIdx.x; c < H_DIM; c += (int)blockDim.x)
    in[c] = agg[(long)n * H_DIM + c] + b2[c];
  __syncthreads();
  for (int o = (int)threadIdx.x; o < OUT_COLS; o += (int)blockDim.x) {
    float acc = bp[o];
#pragma unroll 8
    for (int c = 0; c < H_DIM; ++c) acc += in[c] * Wp[(long)c * OUT_COLS + o];
    out[(long)g * OUT_COLS + o] = acc;
  }
}

// =====================================================================
// host-side launcher
// =====================================================================
static void run_gat_layer(const float* in, int K, const float* W,
                          const float* a_s, const float* a_d, const int* ei,
                          float* hbuf, float* aggbuf,
                          float* alpha_s, float* alpha_d, float* mx, float* den,
                          float* evals, hipStream_t stream) {
  gat_gemm_wmma<<<N_NODES / 16, GEMM_THREADS, 0, stream>>>(in, W, hbuf,
                                                           N_NODES, K, H_DIM);
  gat_node_alpha<<<(N_NODES * 32 + 255) / 256, 256, 0, stream>>>(
      hbuf, a_s, a_d, alpha_s, alpha_d);
  const long tot = (long)N_NODES * H_DIM;
  gat_init_seg<<<(int)((tot + 255) / 256), 256, 0, stream>>>(aggbuf, mx, den);
  gat_edge_logit_max<<<(E_TOT + 255) / 256, 256, 0, stream>>>(ei, alpha_s, alpha_d,
                                                              evals, mx);
  gat_edge_expsum<<<(E_TOT + 255) / 256, 256, 0, stream>>>(ei, evals, mx, den);
  gat_edge_aggregate<<<(int)(((long)E_TOT * 32 + 255) / 256), 256, 0, stream>>>(
      ei, hbuf, evals, den, aggbuf);
}

extern "C" void kernel_launch(void* const* d_in, const int* in_sizes, int n_in,
                              void* d_out, int out_size, void* d_ws, size_t ws_size,
                              hipStream_t stream) {
  // inputs in setup_inputs() order
  const float* x   = (const float*)d_in[0];
  const int*   ei  = (const int*)  d_in[1];
  const int*   bat = (const int*)  d_in[2];
  const float* W0  = (const float*)d_in[3];
  const float* as0 = (const float*)d_in[4];
  const float* ad0 = (const float*)d_in[5];
  const float* b0  = (const float*)d_in[6];
  const float* W1  = (const float*)d_in[7];
  const float* as1 = (const float*)d_in[8];
  const float* ad1 = (const float*)d_in[9];
  const float* b1  = (const float*)d_in[10];
  const float* W2  = (const float*)d_in[11];
  const float* as2 = (const float*)d_in[12];
  const float* ad2 = (const float*)d_in[13];
  const float* b2  = (const float*)d_in[14];
  const float* Wp  = (const float*)d_in[15];
  const float* bp  = (const float*)d_in[16];
  (void)in_sizes; (void)n_in; (void)out_size; (void)ws_size;

  // carve workspace (256B aligned)
  char* w = (char*)d_ws;
  auto carve = [&](size_t bytes) -> void* {
    void* p = (void*)w;
    w += (bytes + 255) & ~(size_t)255;
    return p;
  };
  const size_t HB = sizeof(float) * (size_t)N_NODES * H_DIM;   // 51.2 MB
  float* B0      = (float*)carve(HB);
  float* B1      = (float*)carve(HB);
  float* B2      = (float*)carve(HB);
  float* alpha_s = (float*)carve(sizeof(float) * N_NODES);
  float* alpha_d = (float*)carve(sizeof(float) * N_NODES);
  float* mx      = (float*)carve(sizeof(float) * N_NODES);
  float* den     = (float*)carve(sizeof(float) * N_NODES);
  float* evals   = (float*)carve(sizeof(float) * E_TOT);
  int*   node0   = (int*)  carve(sizeof(int) * B_GRAPHS);

  const long tot = (long)N_NODES * H_DIM;
  const int  ep_blocks = (int)((tot + 255) / 256);

  // layer 0: in=x(K=128) -> h=B0, agg=B1, act0 -> B0
  run_gat_layer(x, D_IN, W0, as0, ad0, ei, B0, B1,
                alpha_s, alpha_d, mx, den, evals, stream);
  gat_bias_relu<<<ep_blocks, 256, 0, stream>>>(B1, b0, B0);

  // layer 1: in=B0 -> h=B1, agg=B2, act1 -> B1
  run_gat_layer(B0, H_DIM, W1, as1, ad1, ei, B1, B2,
                alpha_s, alpha_d, mx, den, evals, stream);
  gat_bias_relu<<<ep_blocks, 256, 0, stream>>>(B2, b1, B1);

  // layer 2 (no relu): in=B1 -> h=B2, agg=B0
  run_gat_layer(B1, H_DIM, W2, as2, ad2, ei, B2, B0,
                alpha_s, alpha_d, mx, den, evals, stream);

  // readout
  gat_find_node0<<<1, 64, 0, stream>>>(bat, node0);
  gat_project<<<B_GRAPHS, 256, 0, stream>>>(B0, b2, node0, Wp, bp, (float*)d_out);
}
